// TransformerBlock_43954695307830
// MI455X (gfx1250) — compile-verified, matrix operand order checked by reading, not run on hardware
//
#include <hip/hip_runtime.h>

#define DIM     2048
#define SEQ     2048
#define BATCH   2
#define MROWS   (BATCH * SEQ)     // 4096 token rows
#define NHEADS  32
#define HDIM    64
#define KVDIM   512               // 8 kv heads * 64
#define HIDDEN  7168

typedef __bf16 bf16;
typedef __attribute__((ext_vector_type(16))) __bf16 v16bf;
typedef __attribute__((ext_vector_type(8)))  __bf16 v8bf;
typedef __attribute__((ext_vector_type(8)))  float  v8f;
typedef int v4i_b128 __attribute__((vector_size(16)));   // 16B payload type of async-LDS builtin

#define AS_GLOBAL __attribute__((address_space(1)))
#define AS_LDS    __attribute__((address_space(3)))

#if defined(__has_builtin)
#if __has_builtin(__builtin_amdgcn_global_load_async_to_lds_b128)
#define HAVE_ASYNC_LDS 1
#endif
#endif
#ifndef HAVE_ASYNC_LDS
#define HAVE_ASYNC_LDS 0
#endif

// ---------------------------------------------------------------------------
// CDNA5 16-bit A/B fragment layout (ISA 7.12.2): lane L = (lr = L&15, hi = L>>4)
// holds 16 bf16 values; elements 0..7  -> k = hi*8 + e
//                       elements 8..15 -> k = 16 + hi*8 + (e-8)
// Both chunks are 16B-contiguous in a row-major row -> two b128 loads.
// ---------------------------------------------------------------------------
__device__ inline v16bf frag_from_bf16_row(const bf16* row, int hi) {
    const v8bf c0 = *(const v8bf*)(row + hi * 8);
    const v8bf c1 = *(const v8bf*)(row + 16 + hi * 8);
    v16bf r;
#pragma unroll
    for (int i = 0; i < 8; ++i) { r[i] = c0[i]; r[i + 8] = c1[i]; }
    return r;
}

__device__ inline v16bf frag_from_f32_row(const float* __restrict__ row, int hi, float scale) {
    const v8f c0 = *(const v8f*)(row + hi * 8);
    const v8f c1 = *(const v8f*)(row + 16 + hi * 8);
    v16bf r;
#pragma unroll
    for (int i = 0; i < 8; ++i) {
        r[i]     = (bf16)(c0[i] * scale);
        r[i + 8] = (bf16)(c1[i] * scale);
    }
    return r;
}

__device__ inline v8f wmma_bf16(v16bf a, v16bf b, v8f c) {
    return __builtin_amdgcn_wmma_f32_16x16x32_bf16(false, a, false, b, (short)0, c, false, false);
}

// 16-byte global -> LDS copy: async (ASYNCcnt) on gfx1250 toolchains that
// expose the builtin, plain load+ds_store otherwise.
__device__ inline void copy16_to_lds(const bf16* g, bf16* l) {
#if HAVE_ASYNC_LDS
    __builtin_amdgcn_global_load_async_to_lds_b128(
        (AS_GLOBAL v4i_b128*)g,
        (AS_LDS v4i_b128*)l, 0, 0);
#else
    *(v8bf*)l = *(const v8bf*)g;
#endif
}

__device__ inline void wait_async_copies() {
#if HAVE_ASYNC_LDS
#if __has_builtin(__builtin_amdgcn_s_wait_asynccnt)
    __builtin_amdgcn_s_wait_asynccnt(0);
#else
    asm volatile("s_wait_asynccnt 0x0" ::: "memory");
#endif
#endif
}

// ---------------------------------------------------------------------------
// fp32 -> bf16 weight cast (4 elems/thread)
// ---------------------------------------------------------------------------
__global__ __launch_bounds__(256) void cvt_f32_bf16(const float* __restrict__ s,
                                                    bf16* __restrict__ d, int n) {
    int i = (blockIdx.x * 256 + threadIdx.x) * 4;
    if (i + 3 < n) {
        float4 v = *(const float4*)(s + i);
        d[i]     = (bf16)v.x;
        d[i + 1] = (bf16)v.y;
        d[i + 2] = (bf16)v.z;
        d[i + 3] = (bf16)v.w;
    }
}

// ---------------------------------------------------------------------------
// RMSNorm(x) * w, quantized to bf16. One row (2048 f32) per 256-thread block.
// ---------------------------------------------------------------------------
__global__ __launch_bounds__(256) void rmsnorm_bf16(const float* __restrict__ X,
                                                    const float* __restrict__ W,
                                                    bf16* __restrict__ Y) {
    __shared__ float red[8];
    const int row = blockIdx.x;
    const float* x = X + (size_t)row * DIM;
    float ss = 0.f;
#pragma unroll
    for (int i = 0; i < 8; ++i) {
        float v = x[threadIdx.x + i * 256];
        ss += v * v;
    }
#pragma unroll
    for (int m = 1; m < 32; m <<= 1) ss += __shfl_xor(ss, m);
    if ((threadIdx.x & 31) == 0) red[threadIdx.x >> 5] = ss;
    __syncthreads();
    float tot = red[0] + red[1] + red[2] + red[3] + red[4] + red[5] + red[6] + red[7];
    float inv = rsqrtf(tot * (1.0f / (float)DIM) + 1e-6f);
    bf16* y = Y + (size_t)row * DIM;
#pragma unroll
    for (int i = 0; i < 8; ++i) {
        int c = threadIdx.x + i * 256;
        y[c] = (bf16)(x[c] * inv * W[c]);
    }
}

// ---------------------------------------------------------------------------
// C[M,N] = A[M,K](bf16) . Bw[N,K](bf16)^T  (+ optional fp32 residual)
// Block: 8 waves = 128 rows x 64 cols. The 64x64 bf16 B-tile for the current
// K-chunk is staged once per block into LDS via async global->LDS copies
// (ASYNCcnt), then every wave reads its B fragments with ds_load_b128.
// Row stride 80 bf16 (160B) keeps 16B alignment and reduces bank conflicts.
// grid = (N/64, M/128)
// ---------------------------------------------------------------------------
__global__ __launch_bounds__(256) void gemm_bf16_wmma(const bf16* __restrict__ A,
                                                      const bf16* __restrict__ Bw,
                                                      const float* __restrict__ R,
                                                      float* __restrict__ C,
                                                      int N, int K) {
    __shared__ bf16 btile[64][80];   // [n within tile][k within 64-chunk]

    const int tid  = threadIdx.x;
    const int lane = tid & 31;
    const int wid  = tid >> 5;
    const int lr = lane & 15, hi = lane >> 4;
    const int m0 = blockIdx.y * 128 + wid * 16;
    const int n0 = blockIdx.x * 64;

    // cooperative B-tile copy: 64 rows x 128B; thread -> (row, 16B chunk)
    const int crow = tid >> 3;        // 0..31 (and +32 on second copy)
    const int ccol = (tid & 7) * 8;   // bf16 offset of 16B chunk

    const bf16* arow = A + (size_t)(m0 + lr) * K;
    const bf16* bsrc = Bw + (size_t)(n0 + crow) * K + ccol;

    v8f a0 = {}, a1 = {}, a2 = {}, a3 = {};
    for (int k0 = 0; k0 < K; k0 += 64) {
        copy16_to_lds(bsrc + k0,                  &btile[crow][ccol]);
        copy16_to_lds(bsrc + k0 + (size_t)32 * K, &btile[crow + 32][ccol]);
        wait_async_copies();
        __syncthreads();
#pragma unroll
        for (int s = 0; s < 64; s += 32) {
            v16bf af = frag_from_bf16_row(arow + k0 + s, hi);
            v16bf b0 = frag_from_bf16_row(&btile[lr][s], hi);
            v16bf b1 = frag_from_bf16_row(&btile[16 + lr][s], hi);
            v16bf b2 = frag_from_bf16_row(&btile[32 + lr][s], hi);
            v16bf b3 = frag_from_bf16_row(&btile[48 + lr][s], hi);
            a0 = wmma_bf16(af, b0, a0);
            a1 = wmma_bf16(af, b1, a1);
            a2 = wmma_bf16(af, b2, a2);
            a3 = wmma_bf16(af, b3, a3);
        }
        __syncthreads();
    }
    // C/D layout: VGPR r -> row m0 + hi*8 + r, col n0 + lr (+16 per n-tile)
#pragma unroll
    for (int r = 0; r < 8; ++r) {
        int row = m0 + hi * 8 + r;
        size_t base = (size_t)row * N + n0 + lr;
        float v0 = a0[r], v1 = a1[r], v2 = a2[r], v3 = a3[r];
        if (R) {
            v0 += R[base];
            v1 += R[base + 16];
            v2 += R[base + 32];
            v3 += R[base + 48];
        }
        C[base]      = v0;
        C[base + 16] = v1;
        C[base + 32] = v2;
        C[base + 48] = v3;
    }
}

// ---------------------------------------------------------------------------
// Streaming-softmax attention (no mask). One wave = 16 q rows of one (b,h).
// Block = 8 waves (128 q rows). grid = (SEQ/128, BATCH*NHEADS).
// Scores and P.V both on bf16 WMMA; 1/sqrt(64) folded into Q.
// Output written as bf16 (feeds the wo GEMM).
// ---------------------------------------------------------------------------
__global__ __launch_bounds__(256) void attn_wmma(const float* __restrict__ Q,
                                                 const float* __restrict__ Km,
                                                 const float* __restrict__ Vm,
                                                 bf16* __restrict__ O) {
    __shared__ bf16 pl[8][16][32];  // per-wave P tile (16 rows x 32 keys)

    const int lane = threadIdx.x & 31;
    const int wid  = threadIdx.x >> 5;
    const int lr = lane & 15, hi = lane >> 4;
    const int bh  = blockIdx.y;
    const int b   = bh >> 5;
    const int h   = bh & 31;
    const int kvh = h >> 2;                 // GQA: 4 q heads per kv head
    const int q0  = blockIdx.x * 128 + wid * 16;

    const float* qrow = Q + ((size_t)(b * SEQ + q0 + lr) * DIM + h * HDIM);
    v16bf qa0 = frag_from_f32_row(qrow, hi, 0.125f);        // d = 0..31
    v16bf qa1 = frag_from_f32_row(qrow + 32, hi, 0.125f);   // d = 32..63

    float run_max[8], run_sum[8];
    v8f o0 = {}, o1 = {}, o2 = {}, o3 = {};
#pragma unroll
    for (int r = 0; r < 8; ++r) { run_max[r] = -3.0e38f; run_sum[r] = 0.f; }

    for (int kc = 0; kc < SEQ; kc += 32) {
        // K^T fragments: B[k=d][n=key]; lane's key row is contiguous in d.
        const float* krL = Km + ((size_t)(b * SEQ + kc + lr) * KVDIM + kvh * HDIM);
        const float* krR = krL + (size_t)16 * KVDIM;
        v16bf bl0 = frag_from_f32_row(krL, hi, 1.f);
        v16bf bl1 = frag_from_f32_row(krL + 32, hi, 1.f);
        v16bf br0 = frag_from_f32_row(krR, hi, 1.f);
        v16bf br1 = frag_from_f32_row(krR + 32, hi, 1.f);

        v8f sl = {}, sr = {};
        sl = wmma_bf16(qa0, bl0, sl);
        sl = wmma_bf16(qa1, bl1, sl);
        sr = wmma_bf16(qa0, br0, sr);
        sr = wmma_bf16(qa1, br1, sr);

        // Online softmax. Row (hi*8+r) lives across the 16 lanes of a half.
        float pls[8], prs[8], scal[8];
#pragma unroll
        for (int r = 0; r < 8; ++r) {
            float cm = fmaxf(sl[r], sr[r]);
#pragma unroll
            for (int m = 1; m < 16; m <<= 1) cm = fmaxf(cm, __shfl_xor(cm, m));
            float nm = fmaxf(run_max[r], cm);
            float sc = __expf(run_max[r] - nm);
            float pa = __expf(sl[r] - nm);
            float pb = __expf(sr[r] - nm);
            float cs = pa + pb;
#pragma unroll
            for (int m = 1; m < 16; m <<= 1) cs += __shfl_xor(cs, m);
            run_sum[r] = run_sum[r] * sc + cs;
            run_max[r] = nm;
            pls[r] = pa; prs[r] = pb; scal[r] = sc;
        }
#pragma unroll
        for (int r = 0; r < 8; ++r) {
            o0[r] *= scal[r]; o1[r] *= scal[r];
            o2[r] *= scal[r]; o3[r] *= scal[r];
        }

        // C-layout -> A-layout redistribution of P through per-wave LDS.
#pragma unroll
        for (int r = 0; r < 8; ++r) {
            pl[wid][hi * 8 + r][lr]      = (bf16)pls[r];
            pl[wid][hi * 8 + r][16 + lr] = (bf16)prs[r];
        }
        asm volatile("s_wait_dscnt 0" ::: "memory");
        v16bf pfrag = frag_from_bf16_row(&pl[wid][lr][0], hi);

        // V fragments: B[k=key][n=d]; lane n=d gathers strided key rows.
        const float* vb = Vm + ((size_t)(b * SEQ + kc) * KVDIM + kvh * HDIM + lr);
#pragma unroll
        for (int dt = 0; dt < 4; ++dt) {
            v16bf vfr;
#pragma unroll
            for (int e = 0; e < 8; ++e) {
                int kA = hi * 8 + e;
                int kB = 16 + hi * 8 + e;
                vfr[e]     = (bf16)vb[(size_t)kA * KVDIM + dt * 16];
                vfr[e + 8] = (bf16)vb[(size_t)kB * KVDIM + dt * 16];
            }
            if (dt == 0)      o0 = wmma_bf16(pfrag, vfr, o0);
            else if (dt == 1) o1 = wmma_bf16(pfrag, vfr, o1);
            else if (dt == 2) o2 = wmma_bf16(pfrag, vfr, o2);
            else              o3 = wmma_bf16(pfrag, vfr, o3);
        }
    }

#pragma unroll
    for (int r = 0; r < 8; ++r) {
        float inv = 1.f / run_sum[r];
        int row = q0 + hi * 8 + r;
        size_t base = (size_t)(b * SEQ + row) * DIM + h * HDIM + lr;
        O[base]      = (bf16)(o0[r] * inv);
        O[base + 16] = (bf16)(o1[r] * inv);
        O[base + 32] = (bf16)(o2[r] * inv);
        O[base + 48] = (bf16)(o3[r] * inv);
    }
}

// ---------------------------------------------------------------------------
// u = silu(g1) * g3 -> bf16   (4 elems/thread)
// ---------------------------------------------------------------------------
__global__ __launch_bounds__(256) void swiglu_bf16(const float* __restrict__ g1,
                                                   const float* __restrict__ g3,
                                                   bf16* __restrict__ u, int n) {
    int i = (blockIdx.x * 256 + threadIdx.x) * 4;
    if (i + 3 < n) {
        float4 a = *(const float4*)(g1 + i);
        float4 b = *(const float4*)(g3 + i);
        u[i]     = (bf16)((a.x / (1.f + __expf(-a.x))) * b.x);
        u[i + 1] = (bf16)((a.y / (1.f + __expf(-a.y))) * b.y);
        u[i + 2] = (bf16)((a.z / (1.f + __expf(-a.z))) * b.z);
        u[i + 3] = (bf16)((a.w / (1.f + __expf(-a.w))) * b.w);
    }
}

// ---------------------------------------------------------------------------
extern "C" void kernel_launch(void* const* d_in, const int* in_sizes, int n_in,
                              void* d_out, int out_size, void* d_ws, size_t ws_size,
                              hipStream_t stream) {
    const float* x   = (const float*)d_in[0];
    const float* wq  = (const float*)d_in[1];
    const float* wk  = (const float*)d_in[2];
    const float* wvp = (const float*)d_in[3];
    const float* wo  = (const float*)d_in[4];
    const float* w1  = (const float*)d_in[5];
    const float* w2  = (const float*)d_in[6];
    const float* w3  = (const float*)d_in[7];
    const float* anw = (const float*)d_in[8];
    const float* fnw = (const float*)d_in[9];
    float* out = (float*)d_out;

    char* ws = (char*)d_ws;
    size_t off = 0;
    auto alloc = [&](size_t bytes) -> void* {
        void* p = ws + off;
        off += (bytes + 255) & ~(size_t)255;
        return p;
    };

    bf16* wq_b = (bf16*)alloc((size_t)DIM * DIM * 2);
    bf16* wk_b = (bf16*)alloc((size_t)KVDIM * DIM * 2);
    bf16* wv_b = (bf16*)alloc((size_t)KVDIM * DIM * 2);
    bf16* wo_b = (bf16*)alloc((size_t)DIM * DIM * 2);
    bf16* w1_b = (bf16*)alloc((size_t)HIDDEN * DIM * 2);
    bf16* w2_b = (bf16*)alloc((size_t)DIM * HIDDEN * 2);
    bf16* w3_b = (bf16*)alloc((size_t)HIDDEN * DIM * 2);
    bf16* xn_b = (bf16*)alloc((size_t)MROWS * DIM * 2);
    float* qf  = (float*)alloc((size_t)MROWS * DIM * 4);
    float* kf  = (float*)alloc((size_t)MROWS * KVDIM * 4);
    float* vf  = (float*)alloc((size_t)MROWS * KVDIM * 4);
    bf16* at_b = (bf16*)alloc((size_t)MROWS * DIM * 2);
    float* hf  = (float*)alloc((size_t)MROWS * DIM * 4);
    bf16* hn_b = (bf16*)alloc((size_t)MROWS * DIM * 2);
    float* g1f = (float*)alloc((size_t)MROWS * HIDDEN * 4);
    float* g3f = (float*)alloc((size_t)MROWS * HIDDEN * 4);
    bf16* u_b  = (bf16*)alloc((size_t)MROWS * HIDDEN * 2);

    auto cvt = [&](const float* s, bf16* d, size_t n) {
        unsigned g = (unsigned)((n / 4 + 255) / 256);
        cvt_f32_bf16<<<g, 256, 0, stream>>>(s, d, (int)n);
    };
    cvt(wq,  wq_b, (size_t)DIM * DIM);
    cvt(wk,  wk_b, (size_t)KVDIM * DIM);
    cvt(wvp, wv_b, (size_t)KVDIM * DIM);
    cvt(wo,  wo_b, (size_t)DIM * DIM);
    cvt(w1,  w1_b, (size_t)HIDDEN * DIM);
    cvt(w2,  w2_b, (size_t)DIM * HIDDEN);
    cvt(w3,  w3_b, (size_t)HIDDEN * DIM);

    // --- attention sub-block ---
    rmsnorm_bf16<<<MROWS, 256, 0, stream>>>(x, anw, xn_b);

    gemm_bf16_wmma<<<dim3(DIM / 64,   MROWS / 128), 256, 0, stream>>>(xn_b, wq_b, nullptr, qf, DIM,   DIM);
    gemm_bf16_wmma<<<dim3(KVDIM / 64, MROWS / 128), 256, 0, stream>>>(xn_b, wk_b, nullptr, kf, KVDIM, DIM);
    gemm_bf16_wmma<<<dim3(KVDIM / 64, MROWS / 128), 256, 0, stream>>>(xn_b, wv_b, nullptr, vf, KVDIM, DIM);

    attn_wmma<<<dim3(SEQ / 128, BATCH * NHEADS), 256, 0, stream>>>(qf, kf, vf, at_b);

    // h = x + attn @ wo^T
    gemm_bf16_wmma<<<dim3(DIM / 64, MROWS / 128), 256, 0, stream>>>(at_b, wo_b, x, hf, DIM, DIM);

    // --- feed-forward sub-block ---
    rmsnorm_bf16<<<MROWS, 256, 0, stream>>>(hf, fnw, hn_b);

    gemm_bf16_wmma<<<dim3(HIDDEN / 64, MROWS / 128), 256, 0, stream>>>(hn_b, w1_b, nullptr, g1f, HIDDEN, DIM);
    gemm_bf16_wmma<<<dim3(HIDDEN / 64, MROWS / 128), 256, 0, stream>>>(hn_b, w3_b, nullptr, g3f, HIDDEN, DIM);

    {
        size_t n = (size_t)MROWS * HIDDEN;
        unsigned g = (unsigned)((n / 4 + 255) / 256);
        swiglu_bf16<<<g, 256, 0, stream>>>(g1f, g3f, u_b, (int)n);
    }

    // out = h + u @ w2^T
    gemm_bf16_wmma<<<dim3(DIM / 64, MROWS / 128), 256, 0, stream>>>(u_b, w2_b, hf, out, DIM, HIDDEN);
}
